// TemporalAxialAttention_67070209294672
// MI455X (gfx1250) — matrix operand order repaired and checked by reading, not compiled, not run on hardware
//
#include <hip/hip_runtime.h>

// ---------------------------------------------------------------------------
// CDNA5 (gfx1250) temporal axial attention, bandwidth-optimized:
//   1) qkv_f16 = x_f32 @ w_qkv_f32            (f16 WMMA, f32 accum)
//   2) per-(n,h) 64x64 causal attention + fused interleaved RoPE (f16 in/out)
//   3) out_f32 = o_f16 @ w_out_f32 + b_out    (A staged via async-to-LDS)
// Intermediates live in HBM as f16: total traffic ~0.8 GB vs 1.34 GB for f32.
// ---------------------------------------------------------------------------

typedef __attribute__((ext_vector_type(16))) _Float16     v16h;
typedef __attribute__((ext_vector_type(4)))  _Float16     v4h;
typedef __attribute__((ext_vector_type(8)))  float        v8f;
typedef __attribute__((ext_vector_type(8)))  unsigned int v8u;

__device__ __forceinline__ v8f wmma_f16(v16h a, v16h b, v8f c) {
  // (neg_a, A, neg_b, B, c_mod, C, reuse_a, reuse_b)
  return __builtin_amdgcn_wmma_f32_16x16x32_f16(false, a, false, b, (short)0, c,
                                                false, false);
}

// A fragment (16x32 f16): lane m = lane&15, hs = lane>>4.
// v0..7 -> k = hs*16 + 2v (contiguous, matches load_bfrag).
__device__ __forceinline__ v16h load_afrag(const unsigned int* lds, int row,
                                           int rowDwords, int dbase, int hs) {
  v8u au;
  const int base = row * rowDwords + dbase;
#pragma unroll
  for (int v = 0; v < 4; ++v) au[v] = lds[base + hs * 8 + v];
#pragma unroll
  for (int v = 0; v < 4; ++v) au[4 + v] = lds[base + hs * 8 + 4 + v];
  return __builtin_bit_cast(v16h, au);
}

// B fragment (32x16 f16) from a TRANSPOSED store (row = n, k contiguous):
// lane n holds k = hs*16 + 2v .. +1  -> 8 contiguous dwords.
__device__ __forceinline__ v16h load_bfrag(const unsigned int* lds, int row,
                                           int rowDwords, int dbase, int hs) {
  v8u bu;
  const int base = row * rowDwords + dbase + hs * 8;
#pragma unroll
  for (int v = 0; v < 8; ++v) bu[v] = lds[base + v];
  return __builtin_bit_cast(v16h, bu);
}

// Low 32 bits of a flat shared-aperture pointer == LDS byte offset (ISA §10.2).
__device__ __forceinline__ unsigned lds_off(const void* p) {
  return (unsigned)(unsigned long long)p;
}

// ---------------------------------------------------------------------------
// GEMM1: C_f16[M,N] = A_f32[M,K] @ B_f32[K,N].  Tile 128x64, K-chunk 32.
// ---------------------------------------------------------------------------
__global__ __launch_bounds__(256) void gemm_f32a_f16c(
    const float* __restrict__ A, const float* __restrict__ B,
    _Float16* __restrict__ C, int M, int N, int K) {
  __shared__ _Float16 As[128 * 32];  // [row][k]
  __shared__ _Float16 Bt[64 * 32];   // transposed [n][k]

  const int tid  = threadIdx.x;
  const int lane = tid & 31;
  const int wave = tid >> 5;
  const int m16  = lane & 15;
  const int hs   = lane >> 4;
  const int blockM = blockIdx.y * 128;
  const int blockN = blockIdx.x * 64;

  v8f acc[4] = {};

  for (int kb = 0; kb < K; kb += 32) {
#pragma unroll
    for (int i = 0; i < 4; ++i) {  // A: 128x32 f32 -> f16
      const int lin = tid + i * 256;
      const int row = lin >> 3;
      const int c4  = (lin & 7) << 2;
      const float4 av =
          *(const float4*)(A + (size_t)(blockM + row) * K + kb + c4);
      As[row * 32 + c4 + 0] = (_Float16)av.x;
      As[row * 32 + c4 + 1] = (_Float16)av.y;
      As[row * 32 + c4 + 2] = (_Float16)av.z;
      As[row * 32 + c4 + 3] = (_Float16)av.w;
    }
#pragma unroll
    for (int i = 0; i < 2; ++i) {  // B: 32x64 f32 -> Bt[n][k] f16
      const int lin = tid + i * 256;
      const int kr  = lin >> 4;
      const int c4  = (lin & 15) << 2;
      const float4 bv =
          *(const float4*)(B + (size_t)(kb + kr) * N + blockN + c4);
      Bt[(c4 + 0) * 32 + kr] = (_Float16)bv.x;
      Bt[(c4 + 1) * 32 + kr] = (_Float16)bv.y;
      Bt[(c4 + 2) * 32 + kr] = (_Float16)bv.z;
      Bt[(c4 + 3) * 32 + kr] = (_Float16)bv.w;
    }
    __syncthreads();

    const unsigned int* Asu = (const unsigned int*)As;
    const unsigned int* Btu = (const unsigned int*)Bt;
    const v16h af = load_afrag(Asu, wave * 16 + m16, 16, 0, hs);
#pragma unroll
    for (int j = 0; j < 4; ++j) {
      const v16h bf = load_bfrag(Btu, j * 16 + m16, 16, 0, hs);
      acc[j] = wmma_f16(af, bf, acc[j]);
    }
    __syncthreads();
  }

#pragma unroll
  for (int j = 0; j < 4; ++j) {
    const int col = blockN + j * 16 + m16;
#pragma unroll
    for (int r = 0; r < 8; ++r) {
      const int row = blockM + wave * 16 + r + 8 * hs;
      C[(size_t)row * N + col] = (_Float16)acc[j][r];
    }
  }
}

// ---------------------------------------------------------------------------
// GEMM2: C_f32[M,N] = A_f16[M,K] @ B_f32[K,N] + bias.
// A tile is a pure f16 copy -> staged with GLOBAL_LOAD_ASYNC_TO_LDS_B128.
// ---------------------------------------------------------------------------
__global__ __launch_bounds__(256) void gemm_f16a_f32c_bias(
    const _Float16* __restrict__ A, const float* __restrict__ B,
    float* __restrict__ C, const float* __restrict__ bias, int M, int N,
    int K) {
  __shared__ _Float16 As[128 * 32];  // [row][k]
  __shared__ _Float16 Bt[64 * 32];   // transposed [n][k]

  const int tid  = threadIdx.x;
  const int lane = tid & 31;
  const int wave = tid >> 5;
  const int m16  = lane & 15;
  const int hs   = lane >> 4;
  const int blockM = blockIdx.y * 128;
  const int blockN = blockIdx.x * 64;

  v8f acc[4] = {};

  for (int kb = 0; kb < K; kb += 32) {
    // A: 128 rows x 64B, async DMA to LDS, 512 x b128 (2 per thread).
#pragma unroll
    for (int i = 0; i < 2; ++i) {
      const int lin = tid + i * 256;
      const int row = lin >> 2;
      const int seg = lin & 3;  // 16B segment within the 64B row chunk
      const unsigned long long gp = (unsigned long long)(
          A + (size_t)(blockM + row) * K + kb + seg * 8);
      const unsigned ldst = lds_off(&As[row * 32 + seg * 8]);
      asm volatile("global_load_async_to_lds_b128 %0, %1, off"
                   :
                   : "v"(ldst), "v"(gp)
                   : "memory");
    }
    // B: 32x64 f32 -> Bt[n][k] f16 (needs convert+transpose: VALU path)
#pragma unroll
    for (int i = 0; i < 2; ++i) {
      const int lin = tid + i * 256;
      const int kr  = lin >> 4;
      const int c4  = (lin & 15) << 2;
      const float4 bv =
          *(const float4*)(B + (size_t)(kb + kr) * N + blockN + c4);
      Bt[(c4 + 0) * 32 + kr] = (_Float16)bv.x;
      Bt[(c4 + 1) * 32 + kr] = (_Float16)bv.y;
      Bt[(c4 + 2) * 32 + kr] = (_Float16)bv.z;
      Bt[(c4 + 3) * 32 + kr] = (_Float16)bv.w;
    }
    asm volatile("s_wait_asynccnt 0x0" ::: "memory");
    __syncthreads();

    const unsigned int* Asu = (const unsigned int*)As;
    const unsigned int* Btu = (const unsigned int*)Bt;
    const v16h af = load_afrag(Asu, wave * 16 + m16, 16, 0, hs);
#pragma unroll
    for (int j = 0; j < 4; ++j) {
      const v16h bf = load_bfrag(Btu, j * 16 + m16, 16, 0, hs);
      acc[j] = wmma_f16(af, bf, acc[j]);
    }
    __syncthreads();
  }

#pragma unroll
  for (int j = 0; j < 4; ++j) {
    const int col = blockN + j * 16 + m16;
    const float bv = bias[col];
#pragma unroll
    for (int r = 0; r < 8; ++r) {
      const int row = blockM + wave * 16 + r + 8 * hs;
      C[(size_t)row * N + col] = acc[j][r] + bv;
    }
  }
}

// ---------------------------------------------------------------------------
// Per-(n,h) causal attention, T=64, d=64, fused interleaved RoPE, f16 HBM I/O.
// 128 threads = 4 waves; wave i owns query rows [i*16, i*16+16).
// ---------------------------------------------------------------------------
__global__ __launch_bounds__(128) void attn_rope_wmma(
    const _Float16* __restrict__ qkv, _Float16* __restrict__ o,
    const float* __restrict__ inv_freq) {
  __shared__ _Float16 Qs[64 * 64];  // [t][d]  (A for Q@K^T)
  __shared__ _Float16 Ks[64 * 64];  // [s][d]  (== B^T for Q@K^T)
  __shared__ _Float16 Vt[64 * 64];  // [d][s]  (== B^T for P@V)
  __shared__ _Float16 Ps[64 * 64];  // [t][s]  probs (A for P@V)

  const int tid  = threadIdx.x;
  const int lane = tid & 31;
  const int wave = tid >> 5;
  const int n16  = lane & 15;
  const int hs   = lane >> 4;

  const int h  = blockIdx.x & 3;
  const int n  = blockIdx.x >> 2;  // (b*32 + y)*32 + w
  const int b  = n >> 10;
  const int yw = n & 1023;
  const size_t rowbase = (size_t)b * 65536 + yw;  // + t*1024 = row in [M]

  // Stage Q,K (RoPE in f32) and V (transposed): 64 rows x 16 v4h each tensor.
#pragma unroll
  for (int i = 0; i < 8; ++i) {
    const int lin = tid + i * 128;  // 0..1023
    const int t   = lin >> 4;
    const int c4  = (lin & 15) << 2;  // d = c4..c4+3
    const size_t roff = (rowbase + (size_t)t * 1024) * 768 + h * 64 + c4;

    const float f0 = inv_freq[c4 >> 1];
    const float f1 = inv_freq[(c4 >> 1) + 1];
    float s0, c0, s1, c1;
    __sincosf((float)t * f0, &s0, &c0);
    __sincosf((float)t * f1, &s1, &c1);

    const v4h qv = *(const v4h*)(qkv + roff);
    const float qx = (float)qv[0], qy = (float)qv[1];
    const float qz = (float)qv[2], qw = (float)qv[3];
    Qs[t * 64 + c4 + 0] = (_Float16)(qx * c0 - qy * s0);
    Qs[t * 64 + c4 + 1] = (_Float16)(qy * c0 + qx * s0);
    Qs[t * 64 + c4 + 2] = (_Float16)(qz * c1 - qw * s1);
    Qs[t * 64 + c4 + 3] = (_Float16)(qw * c1 + qz * s1);

    const v4h kv = *(const v4h*)(qkv + roff + 256);
    const float kx = (float)kv[0], ky = (float)kv[1];
    const float kz = (float)kv[2], kw = (float)kv[3];
    Ks[t * 64 + c4 + 0] = (_Float16)(kx * c0 - ky * s0);
    Ks[t * 64 + c4 + 1] = (_Float16)(ky * c0 + kx * s0);
    Ks[t * 64 + c4 + 2] = (_Float16)(kz * c1 - kw * s1);
    Ks[t * 64 + c4 + 3] = (_Float16)(kw * c1 + kz * s1);

    const v4h vv = *(const v4h*)(qkv + roff + 512);
    Vt[(c4 + 0) * 64 + t] = vv[0];
    Vt[(c4 + 1) * 64 + t] = vv[1];
    Vt[(c4 + 2) * 64 + t] = vv[2];
    Vt[(c4 + 3) * 64 + t] = vv[3];
  }
  __syncthreads();

  // ---- S = Q @ K^T ----
  const unsigned int* Qsu = (const unsigned int*)Qs;
  const unsigned int* Ksu = (const unsigned int*)Ks;
  v8f sacc[4] = {};
#pragma unroll
  for (int c = 0; c < 2; ++c) {  // d-chunks of 32
    const v16h af = load_afrag(Qsu, wave * 16 + n16, 32, c * 16, hs);
#pragma unroll
    for (int j = 0; j < 4; ++j) {
      const v16h bf = load_bfrag(Ksu, j * 16 + n16, 32, c * 16, hs);
      sacc[j] = wmma_f16(af, bf, sacc[j]);
    }
  }

  // ---- causal mask + softmax (row r lives in one VGPR across a 16-lane half)
  const float scale = 0.125f;  // 1/sqrt(64)
  float prob[4][8];
#pragma unroll
  for (int r = 0; r < 8; ++r) {
    const int t_glob = wave * 16 + r + 8 * hs;
    float mx = -3.0e38f;
#pragma unroll
    for (int j = 0; j < 4; ++j) {
      const int s_glob = j * 16 + n16;
      float v = sacc[j][r] * scale;
      if (s_glob > t_glob) v = -3.0e38f;
      prob[j][r] = v;
      mx = fmaxf(mx, v);
    }
#pragma unroll
    for (int off = 1; off < 16; off <<= 1)
      mx = fmaxf(mx, __shfl_xor(mx, off, 32));
    float sum = 0.0f;
#pragma unroll
    for (int j = 0; j < 4; ++j) {
      const float e = __expf(prob[j][r] - mx);
      prob[j][r] = e;
      sum += e;
    }
#pragma unroll
    for (int off = 1; off < 16; off <<= 1) sum += __shfl_xor(sum, off, 32);
    const float inv = 1.0f / sum;
#pragma unroll
    for (int j = 0; j < 4; ++j)
      Ps[(wave * 16 + r + 8 * hs) * 64 + j * 16 + n16] =
          (_Float16)(prob[j][r] * inv);
  }
  __syncthreads();

  // ---- O = P @ V ----
  const unsigned int* Psu = (const unsigned int*)Ps;
  const unsigned int* Vtu = (const unsigned int*)Vt;
  v8f oacc[4] = {};
#pragma unroll
  for (int c = 0; c < 2; ++c) {  // s-chunks of 32
    const v16h af = load_afrag(Psu, wave * 16 + n16, 32, c * 16, hs);
#pragma unroll
    for (int j = 0; j < 4; ++j) {
      const v16h bf = load_bfrag(Vtu, j * 16 + n16, 32, c * 16, hs);
      oacc[j] = wmma_f16(af, bf, oacc[j]);
    }
  }

  // Write O (f16) in (B,T,H,W, h*d) row layout for the output projection.
#pragma unroll
  for (int j = 0; j < 4; ++j) {
    const int dcol = j * 16 + n16;
#pragma unroll
    for (int r = 0; r < 8; ++r) {
      const int t = wave * 16 + r + 8 * hs;
      o[(rowbase + (size_t)t * 1024) * 256 + h * 64 + dcol] =
          (_Float16)oacc[j][r];
    }
  }
}

// ---------------------------------------------------------------------------
extern "C" void kernel_launch(void* const* d_in, const int* in_sizes, int n_in,
                              void* d_out, int out_size, void* d_ws,
                              size_t ws_size, hipStream_t stream) {
  const float* x        = (const float*)d_in[0];  // (2,64,32,32,256)
  const float* w_qkv    = (const float*)d_in[1];  // (256,768)
  const float* w_out    = (const float*)d_in[2];  // (256,256)
  const float* b_out    = (const float*)d_in[3];  // (256,)
  const float* inv_freq = (const float*)d_in[4];  // (32,)
  float* out = (float*)d_out;

  const int M = 2 * 64 * 32 * 32;  // 131072 token rows
  _Float16* qkv   = (_Float16*)d_ws;             // M x 768 f16  (192 MiB)
  _Float16* oattn = qkv + (size_t)M * 768;       // M x 256 f16  ( 64 MiB)

  // 1) QKV projection (f32 in, f16 out)
  gemm_f32a_f16c<<<dim3(768 / 64, M / 128), 256, 0, stream>>>(
      x, w_qkv, qkv, M, 768, 256);

  // 2) fused RoPE + causal attention per (batch*H*W, head)
  attn_rope_wmma<<<dim3(8192), 128, 0, stream>>>(qkv, oattn, inv_freq);

  // 3) output projection + bias (f16 A via async-to-LDS, f32 out)
  gemm_f16a_f32c_bias<<<dim3(256 / 64, M / 128), 256, 0, stream>>>(
      oattn, w_out, out, b_out, M, 256, 256);
}
